// MRIPhysicalLayer_20564303414010
// MI455X (gfx1250) — matrix operand order repaired and checked by reading, not compiled, not run on hardware
//
#include <hip/hip_runtime.h>
#include <math.h>

typedef _Float16 f16;
typedef __attribute__((ext_vector_type(16))) _Float16 v16h;
typedef __attribute__((ext_vector_type(2)))  _Float16 v2h;
typedef __attribute__((ext_vector_type(8)))  float    v8f;

#define NB 256            // B*C batches
#define MU 208            // U
#define NV 176            // V
#define KPU 224           // 208 padded to mult of 32
#define KPV 192           // 176 padded to mult of 32
#define RN  (MU * NV)     // 36608 mask elements
#define RANK_TARGET 18304 // round-half-even((36608-1)*0.5)

// ---------------------------------------------------------------------------
// WMMA helper (CDNA5 wave32, V_WMMA_F32_16X16X32_F16)
// ---------------------------------------------------------------------------
__device__ __forceinline__ v8f wmma_f16(v16h a, v16h b, v8f c) {
  return __builtin_amdgcn_wmma_f32_16x16x32_f16(false, a, false, b, (short)0, c,
                                                false, false);
}

// A fragment: 16x32 f16, row M = lane%16, K layout per ISA:
//   lanes 0-15 : VGPR v<4 -> K=2v,2v+1 ; v>=4 -> K=2v+8   (K in {0..7,16..23})
//   lanes16-31 : +8                                        (K in {8..15,24..31})
// Memory layout: base[row * stride + k], k contiguous.
__device__ __forceinline__ v16h load_fragA(const f16* __restrict__ base,
                                           int stride, int lane, int row0, int kk) {
  const f16* p = base + (size_t)(row0 + (lane & 15)) * stride + kk
                      + ((lane >> 4) << 3);
  v16h f;
#pragma unroll
  for (int v = 0; v < 8; ++v) {
    const int kb = (v < 4) ? (2 * v) : (2 * v + 8);
    v2h w = *reinterpret_cast<const v2h*>(p + kb);
    f[2 * v] = w[0];
    f[2 * v + 1] = w[1];
  }
  return f;
}

// B fragment: 32x16 f16, col N = lane%16, same K layout as A fragment:
// lanes 0-15 K in {0..7,16..23}, lanes 16-31 K in {8..15,24..31}.  Memory: base[col * stride + k].
__device__ __forceinline__ v16h load_fragB(const f16* __restrict__ base,
                                           int stride, int lane, int row0, int kk) {
  const f16* p = base + (size_t)(row0 + (lane & 15)) * stride + kk
                      + ((lane >> 4) << 3);
  v16h f;
#pragma unroll
  for (int v = 0; v < 8; ++v) {
    v2h w = *reinterpret_cast<const v2h*>(p + ((v < 4) ? (2 * v) : (2 * v + 8)));
    f[2 * v] = w[0];
    f[2 * v + 1] = w[1];
  }
  return f;
}

// ---------------------------------------------------------------------------
// Generic complex GEMM stage, 2 batches per wave (twiddle fragments amortized).
//   Y[m][n] = sum_k Aop[m,k] * Bop[n,k]  (both operands K-innermost)
// One side is the complex twiddle T = C + i*SGN*S (C,S split hi/lo f16);
// the other side is per-batch complex data Z = Zr + i*Zi (f16).
//   Yr = C.Zr - SGN*S.Zi ;  Yi = SGN*S.Zr + C.Zi
// TWA: twiddle is the A operand (left DFT); else B operand (right DFT).
// ---------------------------------------------------------------------------
template <int SGN, bool TWA, bool HASZI, bool MASKED, bool TRANSOUT, bool ABSOUT,
          int KTOT>
__global__ __launch_bounds__(32) void cgemm_ker(
    const f16* __restrict__ Tch, const f16* __restrict__ Tcl,
    const f16* __restrict__ Tsh, const f16* __restrict__ Tsl,
    const f16* __restrict__ Zr, const f16* __restrict__ Zi,
    f16* __restrict__ Yr, f16* __restrict__ Yi,
    const float* __restrict__ mask, float* __restrict__ Oabs,
    int strideT, int strideZ, int strideY, size_t planeZ, size_t planeY) {
  const int mt = blockIdx.x, nt = blockIdx.y;
  const int b0 = blockIdx.z * 2;  // two batches per wave
  const int lane = threadIdx.x;

  const f16* zr[2] = {Zr + (size_t)b0 * planeZ, Zr + (size_t)(b0 + 1) * planeZ};
  const f16* zi[2] = {HASZI ? Zi + (size_t)b0 * planeZ : nullptr,
                      HASZI ? Zi + (size_t)(b0 + 1) * planeZ : nullptr};

  v8f aCr[2], aCi[2], aSr[2], aSi[2];
#pragma unroll
  for (int bi = 0; bi < 2; ++bi) {
    aCr[bi] = (v8f){0.f, 0.f, 0.f, 0.f, 0.f, 0.f, 0.f, 0.f};
    aCi[bi] = (v8f){0.f, 0.f, 0.f, 0.f, 0.f, 0.f, 0.f, 0.f};
    aSr[bi] = (v8f){0.f, 0.f, 0.f, 0.f, 0.f, 0.f, 0.f, 0.f};
    aSi[bi] = (v8f){0.f, 0.f, 0.f, 0.f, 0.f, 0.f, 0.f, 0.f};
  }

#pragma unroll
  for (int kk = 0; kk < KTOT; kk += 32) {
    if constexpr (TWA) {
      const v16h ach = load_fragA(Tch, strideT, lane, mt * 16, kk);
      const v16h acl = load_fragA(Tcl, strideT, lane, mt * 16, kk);
      const v16h ash = load_fragA(Tsh, strideT, lane, mt * 16, kk);
      const v16h asl = load_fragA(Tsl, strideT, lane, mt * 16, kk);
#pragma unroll
      for (int bi = 0; bi < 2; ++bi) {
        const v16h bzr = load_fragB(zr[bi], strideZ, lane, nt * 16, kk);
        aCr[bi] = wmma_f16(ach, bzr, aCr[bi]);
        aCr[bi] = wmma_f16(acl, bzr, aCr[bi]);
        aSr[bi] = wmma_f16(ash, bzr, aSr[bi]);
        aSr[bi] = wmma_f16(asl, bzr, aSr[bi]);
        if constexpr (HASZI) {
          const v16h bzi = load_fragB(zi[bi], strideZ, lane, nt * 16, kk);
          aCi[bi] = wmma_f16(ach, bzi, aCi[bi]);
          aCi[bi] = wmma_f16(acl, bzi, aCi[bi]);
          aSi[bi] = wmma_f16(ash, bzi, aSi[bi]);
          aSi[bi] = wmma_f16(asl, bzi, aSi[bi]);
        }
      }
    } else {
      const v16h bch = load_fragB(Tch, strideT, lane, nt * 16, kk);
      const v16h bcl = load_fragB(Tcl, strideT, lane, nt * 16, kk);
      const v16h bsh = load_fragB(Tsh, strideT, lane, nt * 16, kk);
      const v16h bsl = load_fragB(Tsl, strideT, lane, nt * 16, kk);
#pragma unroll
      for (int bi = 0; bi < 2; ++bi) {
        const v16h azr = load_fragA(zr[bi], strideZ, lane, mt * 16, kk);
        aCr[bi] = wmma_f16(azr, bch, aCr[bi]);
        aCr[bi] = wmma_f16(azr, bcl, aCr[bi]);
        aSr[bi] = wmma_f16(azr, bsh, aSr[bi]);
        aSr[bi] = wmma_f16(azr, bsl, aSr[bi]);
        if constexpr (HASZI) {
          const v16h azi = load_fragA(zi[bi], strideZ, lane, mt * 16, kk);
          aCi[bi] = wmma_f16(azi, bch, aCi[bi]);
          aCi[bi] = wmma_f16(azi, bcl, aCi[bi]);
          aSi[bi] = wmma_f16(azi, bsh, aSi[bi]);
          aSi[bi] = wmma_f16(azi, bsl, aSi[bi]);
        }
      }
    }
  }

  const float sgn = (float)SGN;
  const float inv = 1.0f / ((float)MU * (float)NV);
#pragma unroll
  for (int v = 0; v < 8; ++v) {
    const int row = mt * 16 + v + ((lane >> 4) << 3);  // C/D layout: M
    const int col = nt * 16 + (lane & 15);             // C/D layout: N
    float mv = 1.0f;
    if constexpr (MASKED) mv = mask[row * NV + col];
#pragma unroll
    for (int bi = 0; bi < 2; ++bi) {
      const int b = b0 + bi;
      float yr_, yi_;
      if constexpr (HASZI) {
        yr_ = aCr[bi][v] - sgn * aSi[bi][v];
        yi_ = sgn * aSr[bi][v] + aCi[bi][v];
      } else {
        yr_ = aCr[bi][v];
        yi_ = sgn * aSr[bi][v];
      }
      if constexpr (MASKED) {
        yr_ *= mv;
        yi_ *= mv;
      }
      if constexpr (ABSOUT) {
        Oabs[(size_t)b * planeY + (size_t)row * strideY + col] =
            sqrtf(yr_ * yr_ + yi_ * yi_) * inv;
      } else {
        const size_t addr = TRANSOUT ? ((size_t)col * strideY + row)
                                     : ((size_t)row * strideY + col);
        Yr[(size_t)b * planeY + addr] = (f16)yr_;
        Yi[(size_t)b * planeY + addr] = (f16)yi_;
      }
    }
  }
}

// ---------------------------------------------------------------------------
// Precompute padded, hi/lo-split DFT twiddles.
// CU[u][m] = cos(2*pi*u*m/208), SU = sin(...)   (FU = CU - i*SU)
// CV[v][n] = cos(2*pi*v*n/176), SV = sin(...)   (symmetric matrices)
// ---------------------------------------------------------------------------
__global__ void twiddle_ker(f16* CUh, f16* CUl, f16* SUh, f16* SUl, f16* CVh,
                            f16* CVl, f16* SVh, f16* SVl) {
  const int i = blockIdx.x * blockDim.x + threadIdx.x;
  const float TWO_PI = 6.28318530717958647692f;
  if (i < MU * KPU) {
    const int u = i / KPU, m = i % KPU;
    float c = 0.f, s = 0.f;
    if (m < MU) {
      const float ang = TWO_PI * (float)((u * m) % MU) / (float)MU;
      c = cosf(ang);
      s = sinf(ang);
    }
    const f16 ch = (f16)c;
    CUh[i] = ch;
    CUl[i] = (f16)(c - (float)ch);
    const f16 sh = (f16)s;
    SUh[i] = sh;
    SUl[i] = (f16)(s - (float)sh);
  } else if (i < MU * KPU + NV * KPV) {
    const int j = i - MU * KPU;
    const int vv = j / KPV, n = j % KPV;
    float c = 0.f, s = 0.f;
    if (n < NV) {
      const float ang = TWO_PI * (float)((vv * n) % NV) / (float)NV;
      c = cosf(ang);
      s = sinf(ang);
    }
    const f16 ch = (f16)c;
    CVh[j] = ch;
    CVl[j] = (f16)(c - (float)ch);
    const f16 sh = (f16)s;
    SVh[j] = sh;
    SVl[j] = (f16)(s - (float)sh);
  }
}

// x (b,m,n) f32 -> xT (b,n,m-padded) f16 (transposed so stage-1 B fragments
// are contiguous in K=m; pads zero)
__global__ void xprep_ker(const float* __restrict__ x, f16* __restrict__ xT) {
  const size_t i = (size_t)blockIdx.x * blockDim.x + threadIdx.x;
  if (i >= (size_t)NB * NV * KPU) return;
  const int b = (int)(i / (NV * KPU));
  const int rem = (int)(i % (NV * KPU));
  const int n = rem / KPU, m = rem % KPU;
  const float v =
      (m < MU) ? x[(size_t)b * MU * NV + (size_t)m * NV + n] : 0.f;
  xT[i] = (f16)v;
}

// r[l][k] = (1/36) * sum_{m,n<6} sk[m][n] * cos(2*pi*(k*m/176 + l*n/208))
__global__ void r_ker(const float* __restrict__ kker, float* __restrict__ r) {
  const int i = blockIdx.x * blockDim.x + threadIdx.x;
  if (i >= RN) return;
  const float TWO_PI = 6.28318530717958647692f;
  float p[6][6];
#pragma unroll
  for (int m = 0; m < 6; ++m)
#pragma unroll
    for (int n = 0; n < 6; ++n) {
      const int mi = (m < 3) ? m : (5 - m);
      const int ni = (n < 3) ? n : (5 - n);
      p[m][n] = kker[mi * 3 + ni];
    }
  const int l = i / NV, k = i % NV;
  float acc = 0.f;
#pragma unroll
  for (int m = 0; m < 6; ++m)
#pragma unroll
    for (int n = 0; n < 6; ++n) {
      const float ph = (float)((k * m) % NV) / (float)NV +
                       (float)((l * n) % MU) / (float)MU;
      acc += p[m][n] * cosf(TWO_PI * ph);
    }
  r[i] = acc * (1.0f / 36.0f);
}

// rank selection via LDS-tiled counting: find the element whose sorted
// position is RANK_TARGET (numpy percentile method='nearest').
__global__ __launch_bounds__(256) void sel_ker(const float* __restrict__ r,
                                               float* __restrict__ thr) {
  __shared__ float tile[1024];
  const int i = blockIdx.x * blockDim.x + threadIdx.x;
  const float v = (i < RN) ? r[i] : 0.f;
  int less = 0, eq = 0;
  for (int base = 0; base < RN; base += 1024) {
    for (int t = threadIdx.x; t < 1024; t += 256) {
      const int idx = base + t;
      tile[t] = (idx < RN) ? r[idx] : __builtin_inff();  // +inf: counts nothing
    }
    __syncthreads();
#pragma unroll 8
    for (int t = 0; t < 1024; ++t) {
      const float w = tile[t];
      less += (w < v) ? 1 : 0;
      eq += (w == v) ? 1 : 0;
    }
    __syncthreads();
  }
  if (i < RN && less <= RANK_TARGET && RANK_TARGET < less + eq) *thr = v;
}

__global__ void mask_ker(const float* __restrict__ r,
                         const float* __restrict__ thr,
                         float* __restrict__ mask) {
  const int i = blockIdx.x * blockDim.x + threadIdx.x;
  if (i < RN) mask[i] = (r[i] > *thr) ? 1.0f : 0.0f;
}

// ---------------------------------------------------------------------------
extern "C" void kernel_launch(void* const* d_in, const int* in_sizes, int n_in,
                              void* d_out, int out_size, void* d_ws,
                              size_t ws_size, hipStream_t stream) {
  (void)in_sizes; (void)n_in; (void)out_size; (void)ws_size;
  const float* x = (const float*)d_in[0];
  const float* ker = (const float*)d_in[1];
  float* out = (float*)d_out;

  const size_t szTwU = (size_t)MU * KPU;        // 46592
  const size_t szTwV = (size_t)NV * KPV;        // 33792
  const size_t szXT = (size_t)NB * NV * KPU;    // per plane
  const size_t szP1 = (size_t)NB * MU * KPV;
  const size_t szP2 = (size_t)NB * NV * KPU;

  char* ws = (char*)d_ws;
  size_t off = 0;
  auto take = [&](size_t bytes) -> char* {
    char* p = ws + off;
    off += (bytes + 255) & ~(size_t)255;
    return p;
  };
  f16* CUh = (f16*)take(szTwU * 2);
  f16* CUl = (f16*)take(szTwU * 2);
  f16* SUh = (f16*)take(szTwU * 2);
  f16* SUl = (f16*)take(szTwU * 2);
  f16* CVh = (f16*)take(szTwV * 2);
  f16* CVl = (f16*)take(szTwV * 2);
  f16* SVh = (f16*)take(szTwV * 2);
  f16* SVl = (f16*)take(szTwV * 2);
  float* rbuf = (float*)take((size_t)RN * 4);
  float* thr = (float*)take(256);
  float* maskb = (float*)take((size_t)RN * 4);
  f16* xT = (f16*)take(szXT * 2);
  f16* P1r = (f16*)take(szP1 * 2);
  f16* P1i = (f16*)take(szP1 * 2);
  f16* P2r = (f16*)take(szP2 * 2);
  f16* P2i = (f16*)take(szP2 * 2);

  // zero workspace (pads of all K-padded planes must be 0)
  hipMemsetAsync(d_ws, 0, off, stream);

  {
    const int tw_total = MU * KPU + NV * KPV;
    twiddle_ker<<<(tw_total + 255) / 256, 256, 0, stream>>>(
        CUh, CUl, SUh, SUl, CVh, CVl, SVh, SVl);
  }
  {
    const size_t n = szXT;
    xprep_ker<<<(unsigned)((n + 255) / 256), 256, 0, stream>>>(x, xT);
  }
  r_ker<<<(RN + 255) / 256, 256, 0, stream>>>(ker, rbuf);
  sel_ker<<<(RN + 255) / 256, 256, 0, stream>>>(rbuf, thr);
  mask_ker<<<(RN + 255) / 256, 256, 0, stream>>>(rbuf, thr, maskb);

  const dim3 g(MU / 16, NV / 16, NB / 2);  // (13, 11, 128), 2 batches/wave

  // Stage 1: A = FU @ x        (FU = C - iS, x real)
  cgemm_ker<-1, true, false, false, false, false, KPU><<<g, 32, 0, stream>>>(
      CUh, CUl, SUh, SUl, xT, nullptr, P1r, P1i, nullptr, nullptr,
      KPU, KPU, KPV, (size_t)NV * KPU, (size_t)MU * KPV);

  // Stage 2: B = (A @ FV) * mask, stored transposed for stage 3
  cgemm_ker<-1, false, true, true, true, false, KPV><<<g, 32, 0, stream>>>(
      CVh, CVl, SVh, SVl, P1r, P1i, P2r, P2i, maskb, nullptr,
      KPV, KPV, KPU, (size_t)MU * KPV, (size_t)NV * KPU);

  // Stage 3: G = conj(FU) @ B  (conj(FU) = C + iS)
  cgemm_ker<1, true, true, false, false, false, KPU><<<g, 32, 0, stream>>>(
      CUh, CUl, SUh, SUl, P2r, P2i, P1r, P1i, nullptr, nullptr,
      KPU, KPU, KPV, (size_t)NV * KPU, (size_t)MU * KPV);

  // Stage 4: out = |G @ conj(FV)| / (U*V)
  cgemm_ker<1, false, true, false, false, true, KPV><<<g, 32, 0, stream>>>(
      CVh, CVl, SVh, SVl, P1r, P1i, nullptr, nullptr, nullptr, out,
      KPV, KPV, NV, (size_t)MU * KPV, (size_t)MU * NV);
}
